// EdgeGRU_Baseline_53171695124543
// MI455X (gfx1250) — matrix operand order repaired and checked by reading, not compiled, not run on hardware
//
#include <hip/hip_runtime.h>
#include <hip/hip_bf16.h>
#include <stdint.h>

typedef __attribute__((ext_vector_type(16))) __bf16 v16bf;
typedef __attribute__((ext_vector_type(8)))  float  v8f;

#define HID       64
#define EDGE_DIM  32
#define N_NODES   50000
#define N_EDGES   800000
#define LN_EPS    1e-5f

// ------------------------------------------------------------------
// Fragment helpers (CDNA5 16-bit WMMA layouts, wave32)
//  A (16xK): lane m = lane&15, half = lane>>4; element e (0..15):
//            k = ktOff + (e/8)*16 + half*8 + (e%8)   -> two 8-elem runs
//  B (Kx16): lane n = lane&15, half = lane>>4; element e:
//            k = ktOff + (e/8)*16 + half*8 + (e%8)                 (pre-packed)
//
// B fragments are stored as two 512B *planes* per 1KB tile:
//   plane0 = first int4 (8 bf16) of every lane, plane1 = second int4.
// 16B/lane stride -> lane L hits banks 4L mod 64: lanes 0-15 cover all
// 64 banks, lanes 16-31 wrap exactly once => conflict-free ds_load_b128.
// (The old 32B/lane layout was 4-way bank conflicted.)
// ------------------------------------------------------------------

__device__ __forceinline__ v16bf load_b_frag(const __bf16* base, int tile, int lane) {
  const int4* p = (const int4*)(base + (size_t)tile * 512);
  union { v16bf v; int4 q[2]; } u;
  u.q[0] = p[lane];        // plane 0: tile*1024B + lane*16B
  u.q[1] = p[32 + lane];   // plane 1: tile*1024B + 512B + lane*16B
  return u.v;
}

__device__ __forceinline__ void cvt8(__bf16* h, float4 a, float4 b) {
  h[0] = (__bf16)a.x; h[1] = (__bf16)a.y; h[2] = (__bf16)a.z; h[3] = (__bf16)a.w;
  h[4] = (__bf16)b.x; h[5] = (__bf16)b.y; h[6] = (__bf16)b.z; h[7] = (__bf16)b.w;
}

__device__ __forceinline__ void cvt8s(__bf16* h, float4 a, float4 b, float s) {
  h[0] = (__bf16)(a.x * s); h[1] = (__bf16)(a.y * s); h[2] = (__bf16)(a.z * s); h[3] = (__bf16)(a.w * s);
  h[4] = (__bf16)(b.x * s); h[5] = (__bf16)(b.y * s); h[6] = (__bf16)(b.z * s); h[7] = (__bf16)(b.w * s);
}

// Concatenated edge feature row: [ x[src](0..63) | x[dst](64..127) | edge_attr(128..159) ]
__device__ __forceinline__ const float* edge_seg(const float* __restrict__ x,
                                                 const float* __restrict__ ea,
                                                 int s, int d, long long e, int k) {
  if (k < 64)  return x + (size_t)s * 64 + k;
  if (k < 128) return x + (size_t)d * 64 + (k - 64);
  return ea + (size_t)e * EDGE_DIM + (k - 128);
}

__device__ __forceinline__ v16bf load_a_edge(const float* __restrict__ x, const float* __restrict__ ea,
                                             int s, int d, long long e, int kt, int half) {
  union { v16bf v; __bf16 h[16]; } u;
  const int k0 = kt * 32 + half * 8;
  const float* p0 = edge_seg(x, ea, s, d, e, k0);
  const float* p1 = edge_seg(x, ea, s, d, e, k0 + 16);
  cvt8(u.h,     *(const float4*)p0, *(const float4*)(p0 + 4));
  cvt8(u.h + 8, *(const float4*)p1, *(const float4*)(p1 + 4));
  return u.v;
}

// A fragment from an LDS bf16 row (16B-aligned runs)
__device__ __forceinline__ v16bf load_a_lds(const __bf16* rowp, int kt, int half) {
  const int k0 = kt * 32 + half * 8;
  union { v16bf v; int4 q[2]; } u;
  u.q[0] = *(const int4*)(rowp + k0);
  u.q[1] = *(const int4*)(rowp + k0 + 16);
  return u.v;
}

// A fragment from the node-sums row, scaled by 1/count (scatter-mean)
__device__ __forceinline__ v16bf load_a_gru(const float* __restrict__ row, float sc, int kt, int half) {
  const int k0 = kt * 32 + half * 8;
  union { v16bf v; __bf16 h[16]; } u;
  cvt8s(u.h,     *(const float4*)(row + k0),      *(const float4*)(row + k0 + 4),  sc);
  cvt8s(u.h + 8, *(const float4*)(row + k0 + 16), *(const float4*)(row + k0 + 20), sc);
  return u.v;
}

// A fragment for classifier: [h_new[src] | h_new[dst]] already bf16, edge_attr tail f32
__device__ __forceinline__ v16bf load_a_cls(const __bf16* __restrict__ hbuf, const float* __restrict__ ea,
                                            int s, int d, long long e, int kt, int half) {
  union { v16bf v; int4 q[2]; __bf16 h[16]; } u;
  const int kbase = kt * 32 + half * 8;
#pragma unroll
  for (int rsel = 0; rsel < 2; ++rsel) {
    const int k = kbase + rsel * 16;
    if (k < 128) {
      const __bf16* p = (k < 64) ? (hbuf + (size_t)s * 64 + k)
                                 : (hbuf + (size_t)d * 64 + (k - 64));
      u.q[rsel] = *(const int4*)p;
    } else {
      const float* p = ea + (size_t)e * EDGE_DIM + (k - 128);
      cvt8(u.h + rsel * 8, *(const float4*)p, *(const float4*)(p + 4));
    }
  }
  return u.v;
}

// ------------------------------------------------------------------
// Async global -> LDS weight staging (CDNA5: GLOBAL_LOAD_ASYNC_TO_LDS_B128,
// tracked by ASYNCcnt). One b128 per lane per iteration, no VGPR staging.
// LDS aperture maps addr[31:0] to the LDS offset, so truncating the generic
// shared pointer gives the LDS byte address for the VDST operand.
// ------------------------------------------------------------------
__device__ __forceinline__ void stage_weights_async(const __bf16* __restrict__ gsrc,
                                                    __bf16* sdst, int bytes) {
  const int nthr = blockDim.x;
  const int n16  = bytes >> 4;
  const uint64_t sbase   = (uint64_t)(uintptr_t)gsrc;
  const uint32_t ldsbase = (uint32_t)(uintptr_t)sdst;
  for (int i = threadIdx.x; i < n16; i += nthr) {
    const uint32_t voff = (uint32_t)i << 4;        // per-lane global byte offset
    const uint32_t lds  = ldsbase + ((uint32_t)i << 4);
    asm volatile("global_load_async_to_lds_b128 %0, %1, %2"
                 :: "v"(lds), "v"(voff), "s"(sbase)
                 : "memory");
  }
  asm volatile("s_wait_asynccnt 0" ::: "memory");
  __syncthreads();
}

// ------------------------------------------------------------------
// Prep: pack weight matrices into per-lane WMMA B fragments (bf16),
// split-plane layout (see load_b_frag).
// ------------------------------------------------------------------
__device__ void pack_b(const float* __restrict__ W, int K, int Nn, __bf16* __restrict__ dst,
                       int nTilesK, int nTilesN, int transpose, int tid, int nthr) {
  const int total = nTilesK * nTilesN * 32;
  for (int idx = tid; idx < total; idx += nthr) {
    const int lane  = idx & 31;
    const int tile  = idx >> 5;
    const int ktile = tile / nTilesN;
    const int ntile = tile - ktile * nTilesN;
    const int half  = lane >> 4;
    const int n     = ntile * 16 + (lane & 15);
    __bf16* o = dst + (size_t)tile * 512;
    for (int e2 = 0; e2 < 16; ++e2) {
      const int k = ktile * 32 + (e2 >> 3) * 16 + half * 8 + (e2 & 7);
      float v = 0.f;
      if (k < K && n < Nn)
        v = transpose ? W[(size_t)n * K + k] : W[(size_t)k * Nn + n];
      // plane = e2/8 (256 bf16 per plane), 8 bf16 per lane within a plane
      o[(e2 >> 3) * 256 + lane * 8 + (e2 & 7)] = (__bf16)v;
    }
  }
}

__global__ void pack_weights_kernel(const float* __restrict__ W1, const float* __restrict__ W2,
                                    const float* __restrict__ Wih, const float* __restrict__ Wc1,
                                    __bf16* fw1, __bf16* fw2, __bf16* fwih, __bf16* fwc1) {
  const int tid = threadIdx.x, nthr = blockDim.x;
  pack_b(W1, 160,  64, fw1, 5,  4, 0, tid, nthr);
  pack_b(W2,  64,  64, fw2, 2,  4, 0, tid, nthr);
  pack_b(Wih, 64, 192, fwih, 2, 12, 1, tid, nthr);  // B = Wih^T : B[k][n] = Wih[n*64+k]
  pack_b(Wc1, 160, 64, fwc1, 5,  4, 0, tid, nthr);
}

__global__ void zero_kernel(float* __restrict__ p, size_t n) {
  size_t i = (size_t)blockIdx.x * blockDim.x + threadIdx.x;
  const size_t stride = (size_t)gridDim.x * blockDim.x;
  for (; i < n; i += stride) p[i] = 0.f;
}

// ------------------------------------------------------------------
// Kernel 1: edge encoder (GEMM+LN+ReLU x2) fused with scatter-add
// 4 waves/block (128 threads); weights async-staged to LDS once/block.
// LDS: 28KB weights + 16KB C tiles + 8KB A2 tiles = 52KB.
// ------------------------------------------------------------------
__global__ __launch_bounds__(128) void encoder_scatter_kernel(
    const float* __restrict__ x, const int* __restrict__ ei, const float* __restrict__ ea,
    const float* __restrict__ b1, const float* __restrict__ g1, const float* __restrict__ be1,
    const float* __restrict__ b2, const float* __restrict__ g2, const float* __restrict__ be2,
    const __bf16* __restrict__ fw1, // fw2 contiguous after fw1 (20 + 8 tiles)
    float* __restrict__ sums, float* __restrict__ cnt) {
  __shared__ __attribute__((aligned(16))) __bf16 sW[28 * 512];     // 28 KB
  __shared__ __attribute__((aligned(16))) float  sC [4][16 * HID]; // 16 KB
  __shared__ __attribute__((aligned(16))) __bf16 sA2[4][16 * HID]; //  8 KB

  stage_weights_async(fw1, sW, 28 * 512 * 2);
  const __bf16* sW1 = sW;
  const __bf16* sW2 = sW + 20 * 512;

  const int wave = threadIdx.x >> 5;
  const int lane = threadIdx.x & 31;
  const int li = lane & 15, half = lane >> 4;
  const long long e0 = ((long long)blockIdx.x * 4 + wave) * 16;
  const long long eg = e0 + li;                         // this lane's gather row
  const int s = ei[eg], d = ei[(long long)N_EDGES + eg];

  float*  C  = sC[wave];
  __bf16* A2 = sA2[wave];

  // ---- GEMM1: [16x160] x [160x64] ----
  v8f acc[4] = {};
#pragma unroll
  for (int kt = 0; kt < 5; ++kt) {
    v16bf a = load_a_edge(x, ea, s, d, eg, kt, half);
#pragma unroll
    for (int nt = 0; nt < 4; ++nt) {
      v16bf b = load_b_frag(sW1, kt * 4 + nt, lane);
      acc[nt] = __builtin_amdgcn_wmma_f32_16x16x32_bf16(false, a, false, b, (short)0, acc[nt], false, false);
    }
  }
  // spill C tile (VGPR r -> row r lanes 0-15, row r+8 lanes 16-31)
#pragma unroll
  for (int nt = 0; nt < 4; ++nt)
#pragma unroll
    for (int r = 0; r < 8; ++r)
      C[(half * 8 + r) * HID + nt * 16 + li] = acc[nt][r];
  __builtin_amdgcn_wave_barrier();

  // ---- LN1 + ReLU -> bf16 A2 (lane handles row li, cols [half*32, half*32+32)) ----
  {
    float vals[32]; float sum = 0.f, sq = 0.f;
#pragma unroll
    for (int c = 0; c < 32; ++c) {
      const int col = half * 32 + c;
      const float v = C[li * HID + col] + b1[col];
      vals[c] = v; sum += v; sq += v * v;
    }
    sum += __shfl_xor(sum, 16, 32);
    sq  += __shfl_xor(sq, 16, 32);
    const float mean = sum * (1.f / HID);
    const float var  = sq * (1.f / HID) - mean * mean;
    const float inv  = rsqrtf(var + LN_EPS);
#pragma unroll
    for (int c = 0; c < 32; ++c) {
      const int col = half * 32 + c;
      const float h = (vals[c] - mean) * inv * g1[col] + be1[col];
      A2[li * HID + col] = (__bf16)fmaxf(h, 0.f);
    }
  }
  __builtin_amdgcn_wave_barrier();

  // ---- GEMM2: [16x64] x [64x64] ----
  v8f acc2[4] = {};
#pragma unroll
  for (int kt = 0; kt < 2; ++kt) {
    v16bf a = load_a_lds(A2 + li * HID, kt, half);
#pragma unroll
    for (int nt = 0; nt < 4; ++nt) {
      v16bf b = load_b_frag(sW2, kt * 4 + nt, lane);
      acc2[nt] = __builtin_amdgcn_wmma_f32_16x16x32_bf16(false, a, false, b, (short)0, acc2[nt], false, false);
    }
  }
#pragma unroll
  for (int nt = 0; nt < 4; ++nt)
#pragma unroll
    for (int r = 0; r < 8; ++r)
      C[(half * 8 + r) * HID + nt * 16 + li] = acc2[nt][r];
  __builtin_amdgcn_wave_barrier();

  // ---- LN2 + ReLU + scatter-add into node sums (segment_sum) ----
  {
    float vals[32]; float sum = 0.f, sq = 0.f;
#pragma unroll
    for (int c = 0; c < 32; ++c) {
      const int col = half * 32 + c;
      const float v = C[li * HID + col] + b2[col];
      vals[c] = v; sum += v; sq += v * v;
    }
    sum += __shfl_xor(sum, 16, 32);
    sq  += __shfl_xor(sq, 16, 32);
    const float mean = sum * (1.f / HID);
    const float var  = sq * (1.f / HID) - mean * mean;
    const float inv  = rsqrtf(var + LN_EPS);
#pragma unroll
    for (int c = 0; c < 32; ++c) {
      const int col = half * 32 + c;
      const float h = fmaxf((vals[c] - mean) * inv * g2[col] + be2[col], 0.f);
      atomicAdd(&sums[(size_t)s * HID + col], h);
    }
    if (half == 0) atomicAdd(&cnt[s], 1.0f);
  }
}

// ------------------------------------------------------------------
// Kernel 2: GRU cell (h_prev = 0): aggr @ Wih^T, gates, h_new -> bf16
// Weights (24KB) async-staged to LDS once per block.
// ------------------------------------------------------------------
__global__ __launch_bounds__(256) void gru_kernel(
    const float* __restrict__ sums, const float* __restrict__ cnt,
    const __bf16* __restrict__ fwih,
    const float* __restrict__ bih, const float* __restrict__ bhh,
    __bf16* __restrict__ hbuf, int nTiles) {
  __shared__ __attribute__((aligned(16))) __bf16 sW[24 * 512];     // 24 KB
  stage_weights_async(fwih, sW, 24 * 512 * 2);

  const int wave = threadIdx.x >> 5;
  const int lane = threadIdx.x & 31;
  const int li = lane & 15, half = lane >> 4;
  const int tile = blockIdx.x * 8 + wave;
  if (tile >= nTiles) return;
  const int n0 = tile * 16;
  const int node = n0 + li;
  const float invc = 1.f / fmaxf(cnt[node], 1.f);

  v8f acc[12] = {};
#pragma unroll
  for (int kt = 0; kt < 2; ++kt) {
    v16bf a = load_a_gru(sums + (size_t)node * HID, invc, kt, half);
#pragma unroll
    for (int nt = 0; nt < 12; ++nt) {
      v16bf b = load_b_frag(sW, kt * 12 + nt, lane);
      acc[nt] = __builtin_amdgcn_wmma_f32_16x16x32_bf16(false, a, false, b, (short)0, acc[nt], false, false);
    }
  }
  // gates: gh = bhh (h_prev == 0), h_new = (1-z)*n
#pragma unroll
  for (int t = 0; t < 4; ++t) {
    const int col = t * 16 + li;
    const float br = bih[col]       + bhh[col];
    const float bz = bih[64 + col]  + bhh[64 + col];
    const float bn = bih[128 + col];
    const float hn = bhh[128 + col];
#pragma unroll
    for (int r = 0; r < 8; ++r) {
      const float rg = 1.f / (1.f + __expf(-(acc[t][r]     + br)));
      const float zg = 1.f / (1.f + __expf(-(acc[4 + t][r] + bz)));
      const float ng = tanhf(acc[8 + t][r] + bn + rg * hn);
      const float h  = (1.f - zg) * ng;
      hbuf[(size_t)(n0 + half * 8 + r) * HID + col] = (__bf16)h;
    }
  }
}

// ------------------------------------------------------------------
// Kernel 3: classifier: relu(rep @ Wc1 + bc1) @ Wc2 + bc2
// Weights (20KB) async-staged to LDS once per block.
// ------------------------------------------------------------------
__global__ __launch_bounds__(256) void cls_kernel(
    const __bf16* __restrict__ hbuf, const int* __restrict__ ei, const float* __restrict__ ea,
    const __bf16* __restrict__ fwc1, const float* __restrict__ bc1,
    const float* __restrict__ wc2, const float* __restrict__ bc2,
    float* __restrict__ out) {
  __shared__ __attribute__((aligned(16))) __bf16 sW[20 * 512];     // 20 KB
  stage_weights_async(fwc1, sW, 20 * 512 * 2);

  const int wave = threadIdx.x >> 5;
  const int lane = threadIdx.x & 31;
  const int li = lane & 15, half = lane >> 4;
  const long long e0 = ((long long)blockIdx.x * 8 + wave) * 16;
  const long long eg = e0 + li;
  const int s = ei[eg], d = ei[(long long)N_EDGES + eg];

  v8f acc[4] = {};
#pragma unroll
  for (int kt = 0; kt < 5; ++kt) {
    v16bf a = load_a_cls(hbuf, ea, s, d, eg, kt, half);
#pragma unroll
    for (int nt = 0; nt < 4; ++nt) {
      v16bf b = load_b_frag(sW, kt * 4 + nt, lane);
      acc[nt] = __builtin_amdgcn_wmma_f32_16x16x32_bf16(false, a, false, b, (short)0, acc[nt], false, false);
    }
  }
  float part[8] = {0.f, 0.f, 0.f, 0.f, 0.f, 0.f, 0.f, 0.f};
#pragma unroll
  for (int nt = 0; nt < 4; ++nt) {
    const int col = nt * 16 + li;
    const float bb = bc1[col];
    const float w  = wc2[col];
#pragma unroll
    for (int r = 0; r < 8; ++r)
      part[r] += fmaxf(acc[nt][r] + bb, 0.f) * w;
  }
#pragma unroll
  for (int off = 1; off < 16; off <<= 1)
#pragma unroll
    for (int r = 0; r < 8; ++r)
      part[r] += __shfl_xor(part[r], off, 32);
  if (li == 0) {
    const float bb = bc2[0];
#pragma unroll
    for (int r = 0; r < 8; ++r)
      out[e0 + half * 8 + r] = part[r] + bb;
  }
}

// ------------------------------------------------------------------
extern "C" void kernel_launch(void* const* d_in, const int* in_sizes, int n_in,
                              void* d_out, int out_size, void* d_ws, size_t ws_size,
                              hipStream_t stream) {
  const float* x   = (const float*)d_in[0];
  const int*   ei  = (const int*)  d_in[1];
  const float* ea  = (const float*)d_in[2];
  // d_in[3] = global_node_ids (unused)
  const float* W1  = (const float*)d_in[4];
  const float* b1  = (const float*)d_in[5];
  const float* g1  = (const float*)d_in[6];
  const float* be1 = (const float*)d_in[7];
  const float* W2  = (const float*)d_in[8];
  const float* b2  = (const float*)d_in[9];
  const float* g2  = (const float*)d_in[10];
  const float* be2 = (const float*)d_in[11];
  const float* Wih = (const float*)d_in[12];
  // d_in[13] = Whh (unused: h_prev == 0)
  const float* bih = (const float*)d_in[14];
  const float* bhh = (const float*)d_in[15];
  const float* Wc1 = (const float*)d_in[16];
  const float* bc1 = (const float*)d_in[17];
  const float* Wc2 = (const float*)d_in[18];
  const float* bc2 = (const float*)d_in[19];
  float* out = (float*)d_out;

  // Workspace layout (~19.5 MB total)
  float*  sums = (float*)d_ws;                         // N*64 f32
  float*  cnt  = sums + (size_t)N_NODES * HID;         // N   f32
  __bf16* hbuf = (__bf16*)(cnt + N_NODES);             // N*64 bf16
  __bf16* fw1  = hbuf + (size_t)N_NODES * HID;         // 20 tiles * 512 bf16
  __bf16* fw2  = fw1  + 20 * 512;                      //  8 tiles (contiguous with fw1)
  __bf16* fwih = fw2  +  8 * 512;                      // 24 tiles
  __bf16* fwc1 = fwih + 24 * 512;                      // 20 tiles

  const size_t needed = (size_t)N_NODES * HID * 4 + (size_t)N_NODES * 4
                      + (size_t)N_NODES * HID * 2 + (size_t)72 * 512 * 2;
  if (ws_size < needed) return;  // deterministic: ws_size is fixed per run

  // 1) zero the scatter accumulators (sums + cnt are contiguous)
  zero_kernel<<<512, 256, 0, stream>>>(sums, (size_t)N_NODES * (HID + 1));
  // 2) pack weights into WMMA B fragments (split-plane, bank-conflict-free)
  pack_weights_kernel<<<1, 256, 0, stream>>>(W1, W2, Wih, Wc1, fw1, fw2, fwih, fwc1);
  // 3) encoder + scatter-add  (E/64 = 12500 blocks, 4 waves x 16 edges)
  encoder_scatter_kernel<<<N_EDGES / 64, 128, 0, stream>>>(
      x, ei, ea, b1, g1, be1, b2, g2, be2, fw1, sums, cnt);
  // 4) GRU cell over nodes    (N/16 = 3125 tiles, 8 waves/block)
  const int gruTiles = N_NODES / 16;
  gru_kernel<<<(gruTiles + 7) / 8, 256, 0, stream>>>(sums, cnt, fwih, bih, bhh, hbuf, gruTiles);
  // 5) classifier             (E/128 = 6250 blocks)
  cls_kernel<<<N_EDGES / 128, 256, 0, stream>>>(hbuf, ei, ea, fwc1, bc1, Wc2, bc2, out);
}
